// PointFlowModuleWithMaxAvgpool_704374636870
// MI455X (gfx1250) — matrix operand order repaired and checked by reading, not compiled, not run on hardware
//
#include <hip/hip_runtime.h>
#include <hip/hip_bf16.h>
#include <math.h>

// ---------------------------------------------------------------------------
// PointFlow module for MI455X (gfx1250, wave32).
// Heavy stages (down_h/down_l 1x1 convs, ef1 3x3 conv, attention BMMs) run as
// wave-level bf16 WMMA GEMMs (v_wmma_f32_16x16x32_bf16, f32 accumulate).
//  * ef1: block = 8 waves sharing one out-channel block over 8 pixel tiles;
//    per-tap packed weights staged global->LDS with the gfx1250 async path
//    (global_load_async_to_lds_b128 + s_wait_asynccnt), B frags read via DS.
//  * 1x1 convs: each wave computes a 16x64 tile (4 accumulators) so A frags
//    are loaded once per K-step and reused against 4 B frags.
// Light stages (match conv, ef2, pooling, top-k, point-sample, softmax,
// scatter) are plain VALU kernels -- they are <1% of the FLOPs.
// ---------------------------------------------------------------------------

typedef __attribute__((ext_vector_type(16))) __bf16 v16bf;
typedef __attribute__((ext_vector_type(8)))  __bf16 v8bf;
typedef __attribute__((ext_vector_type(8)))  float  v8f;

#define DEV __device__ __forceinline__

static constexpr int BATCH = 8;
static constexpr int CP    = 256;     // feature channels
static constexpr int DD    = 64;      // embedding channels
static constexpr int HH    = 64;      // high-res side
static constexpr int HWH   = 64 * 64;
static constexpr int HL    = 128;     // low-res side
static constexpr int HWL   = 128 * 128;
static constexpr int PP    = 128;     // edge points
static constexpr size_t FINAL_ELEMS = (size_t)BATCH * CP * HWL; // 33,554,432

// ------------------------------- frag helpers ------------------------------
DEV v16bf zero16() {
    v16bf z;
#pragma unroll
    for (int i = 0; i < 16; ++i) z[i] = (__bf16)0.0f;
    return z;
}

// A-matrix frag (16-bit A 16x32 layout): lane half h:
//   K = 16*h + j (contiguous, matches packed B)
// base points at (pixel_row_NHWC + c0); two contiguous 16-byte loads.
DEV v16bf load_a_frag(const __bf16* __restrict__ base, int half) {
    v8bf lo = *(const v8bf*)(base + half * 16);
    v8bf hi = *(const v8bf*)(base + half * 16 + 8);
    v16bf a;
#pragma unroll
    for (int i = 0; i < 8; ++i) { a[i] = lo[i]; a[i + 8] = hi[i]; }
    return a;
}

// B-matrix frag: per-lane 16 contiguous bf16 from a pre-swizzled buffer
// (global or LDS -- compiler picks global_load_b128 / ds_load_b128).
DEV v16bf load_b_frag(const __bf16* __restrict__ p) {
    v8bf lo = *(const v8bf*)p;
    v8bf hi = *(const v8bf*)(p + 8);
    v16bf b;
#pragma unroll
    for (int i = 0; i < 8; ++i) { b[i] = lo[i]; b[i + 8] = hi[i]; }
    return b;
}

// --------------------------- data prep kernels -----------------------------
// NCHW f32 -> NHWC bf16
__global__ void k_to_nhwc_bf16(const float* __restrict__ in, __bf16* __restrict__ out,
                               int C, int HW, int total) {
    int idx = blockIdx.x * blockDim.x + threadIdx.x;
    if (idx >= total) return;
    int p = idx % HW;
    int c = (idx / HW) % C;
    int n = idx / (HW * C);
    out[((size_t)(n * HW + p)) * C + c] = (__bf16)in[idx];
}

// OIHW f32 weights -> packed per-lane B frags:
// flat = (((t*NB + nb)*KB + kb)*32 + lane)*16 + j,
// value = w[Nout = nb*16 + (lane&15)][K = kb*32 + (lane<16?0:16) + j][tap t]
__global__ void k_pack_w(const float* __restrict__ w, __bf16* __restrict__ out,
                         int O, int I, int taps, int NB, int KB, int total) {
    int flat = blockIdx.x * blockDim.x + threadIdx.x;
    if (flat >= total) return;
    int j    = flat & 15;
    int lane = (flat >> 4) & 31;
    int rest = flat >> 9;
    int kb   = rest % KB; rest /= KB;
    int nb   = rest % NB;
    int t    = rest / NB;
    int K  = kb * 32 + ((lane < 16) ? 0 : 16) + j;
    int Nn = nb * 16 + (lane & 15);
    float v = (K < I && Nn < O) ? w[((size_t)Nn * I + K) * taps + t] : 0.0f;
    out[flat] = (__bf16)v;
}

// ------------------------- WMMA 1x1-conv GEMM ------------------------------
// in: NHWC bf16 (Cin=256), Bp: [4][8][32][16], out: NHWC f32 (Cout=64)+bias.
// One wave = 16x64 output tile: A frag loaded once per K step, 4 B frags.
__global__ void k_gemm_1x1(const __bf16* __restrict__ in, const __bf16* __restrict__ bp,
                           const float* __restrict__ bias, float* __restrict__ out,
                           int mtiles) {
    int gt   = blockIdx.x * blockDim.x + threadIdx.x;
    int mt   = gt >> 5;       // pixel tile (16 pixels), one per wave
    int lane = gt & 31;
    if (mt >= mtiles) return; // uniform per wave
    int lm = lane & 15, half = lane >> 4;
    const __bf16* arow = in + ((size_t)(mt * 16 + lm)) * CP;
    v8f acc[4] = {{}, {}, {}, {}};
#pragma unroll
    for (int kb = 0; kb < 8; ++kb) {
        v16bf a = load_a_frag(arow + kb * 32, half);
#pragma unroll
        for (int nb = 0; nb < 4; ++nb) {
            v16bf b = load_b_frag(bp + ((size_t)((nb * 8 + kb) * 32 + lane)) * 16);
            acc[nb] = __builtin_amdgcn_wmma_f32_16x16x32_bf16(false, a, false, b,
                                                              (short)0, acc[nb], false, false);
        }
    }
#pragma unroll
    for (int nb = 0; nb < 4; ++nb) {
        int ch = nb * 16 + lm;
        float bs = bias[ch];
#pragma unroll
        for (int r = 0; r < 8; ++r) {
            int m = half ? (r + 8) : r;
            out[((size_t)(mt * 16 + m)) * DD + ch] = acc[nb][r] + bs;
        }
    }
}

// -------------------- WMMA 3x3 edge-conv (ef1) + BN + ReLU -----------------
// xedge: NHWC bf16 (C=256, 64x64), Bp: [9][16][8][32][16], out NHWC bf16.
// Block = 8 waves: one out-channel block (nb), 8 pixel tiles. Per tap, the
// 8 KB of packed B for (tap, nb) is staged into LDS with async-to-LDS loads
// and then consumed by all 8 waves via DS reads (8x reuse).
__global__ void k_gemm_ef1(const __bf16* __restrict__ xedge, const __bf16* __restrict__ bp,
                           const float* __restrict__ bn_g, const float* __restrict__ bn_b,
                           const float* __restrict__ bn_rm, const float* __restrict__ bn_rv,
                           __bf16* __restrict__ h1relu) {
    __shared__ __bf16 Bs[8 * 32 * 16];   // one tap: [kb][lane][16] = 8 KB
    int tid  = threadIdx.x;
    int w    = tid >> 5;                 // wave in block: 0..7
    int lane = tid & 31;
    int nb   = blockIdx.x & 15;          // out-channel block
    int mtg  = blockIdx.x >> 4;          // group of 8 pixel tiles
    int mt   = mtg * 8 + w;
    int xt = mt & 3;
    int y  = (mt >> 2) & 63;
    int n  = mt >> 8;
    int x0 = xt << 4;
    int lm = lane & 15, half = lane >> 4;
    int x  = x0 + lm;
    v8f acc = {};
    for (int t = 0; t < 9; ++t) {
        // ---- stage this tap's packed B into LDS (async path, ASYNCcnt) ----
#pragma unroll
        for (int i = tid; i < 512; i += 256) {      // 512 x 16-byte chunks
            int kb2    = i >> 6;
            int r      = i & 63;
            int lane_s = r >> 1;
            int h      = r & 1;
            const __bf16* g = bp +
                ((size_t)(((t * 16 + nb) * 8 + kb2) * 32 + lane_s)) * 16 + h * 8;
            unsigned lds = (unsigned)(size_t)&Bs[(kb2 * 32 + lane_s) * 16 + h * 8];
            asm volatile("global_load_async_to_lds_b128 %0, %1, off"
                         :: "v"(lds), "v"(g) : "memory");
        }
        asm volatile("s_wait_asynccnt 0x0" ::: "memory");
        __syncthreads();                            // B visible to all waves
        int dy = t / 3 - 1, dx = t % 3 - 1;
        int yp = y + dy;
        if (yp >= 0 && yp < HH) {                   // wave-uniform predicate
            int xp = x + dx;
            bool ok = (xp >= 0) && (xp < HH);       // per-lane, loads only
            const __bf16* arow = xedge + ((size_t)((n * HH + yp) * HH + xp)) * CP;
#pragma unroll
            for (int kb = 0; kb < 8; ++kb) {
                v16bf a = ok ? load_a_frag(arow + kb * 32, half) : zero16();
                v16bf b = load_b_frag(&Bs[(kb * 32 + lane) * 16]);
                acc = __builtin_amdgcn_wmma_f32_16x16x32_bf16(false, a, false, b,
                                                              (short)0, acc, false, false);
            }
        }
        __syncthreads();                            // protect Bs for next tap
    }
    int ch = (nb << 4) + lm;
    float sc = bn_g[ch] * rsqrtf(bn_rv[ch] + 1e-5f);
    float sh = bn_b[ch] - bn_rm[ch] * sc;
    size_t pixbase = (size_t)((n * HH + y) * HH + x0);
#pragma unroll
    for (int r = 0; r < 8; ++r) {
        int m = half ? (r + 8) : r;
        float v = fmaxf(acc[r] * sc + sh, 0.0f);
        h1relu[(pixbase + m) * CP + ch] = (__bf16)v;
    }
}

// --------------------------- small VALU kernels ----------------------------
// align_corners bilinear resize on NHWC f32
__global__ void k_upsample_ac(const float* __restrict__ in, float* __restrict__ out,
                              int C, int Hi, int Wi, int Ho, int Wo, int total) {
    int idx = blockIdx.x * blockDim.x + threadIdx.x;
    if (idx >= total) return;
    int c = idx % C;
    int rest = idx / C;
    int ox = rest % Wo; rest /= Wo;
    int oy = rest % Ho;
    int n  = rest / Ho;
    float fy = (Ho > 1) ? oy * (float)(Hi - 1) / (float)(Ho - 1) : 0.0f;
    float fx = (Wo > 1) ? ox * (float)(Wi - 1) / (float)(Wo - 1) : 0.0f;
    int y0 = (int)floorf(fy), x0 = (int)floorf(fx);
    if (y0 > Hi - 1) y0 = Hi - 1;
    if (x0 > Wi - 1) x0 = Wi - 1;
    int y1 = min(y0 + 1, Hi - 1), x1 = min(x0 + 1, Wi - 1);
    float wy = fy - y0, wx = fx - x0;
    const float* I = in + (size_t)n * Hi * Wi * C;
    float v00 = I[((size_t)y0 * Wi + x0) * C + c];
    float v01 = I[((size_t)y0 * Wi + x1) * C + c];
    float v10 = I[((size_t)y1 * Wi + x0) * C + c];
    float v11 = I[((size_t)y1 * Wi + x1) * C + c];
    out[idx] = v00 * (1 - wy) * (1 - wx) + v01 * (1 - wy) * wx +
               v10 * wy * (1 - wx)      + v11 * wy * wx;
}

// 3x3 conv over concat(xh_emb, xl_up) (128ch NHWC) -> sigmoid -> certainty
__global__ void k_match(const float* __restrict__ xh_emb, const float* __restrict__ xl_up,
                        const float* __restrict__ mw, const float* __restrict__ mb,
                        float* __restrict__ cert, int total) {
    int idx = blockIdx.x * blockDim.x + threadIdx.x;
    if (idx >= total) return;
    int n = idx / HWH;
    int pix = idx % HWH;
    int y = pix >> 6, x = pix & 63;
    float acc = mb[0];
    for (int ky = 0; ky < 3; ++ky) {
        int yy = y + ky - 1;
        if (yy < 0 || yy >= HH) continue;
        for (int kx = 0; kx < 3; ++kx) {
            int xx = x + kx - 1;
            if (xx < 0 || xx >= HH) continue;
            const float* ph = xh_emb + ((size_t)(n * HWH + yy * HH + xx)) * DD;
            const float* pl = xl_up + ((size_t)(n * HWH + yy * HH + xx)) * DD;
            for (int c = 0; c < DD; ++c) {
                acc += ph[c] * mw[((size_t)c * 3 + ky) * 3 + kx];
                acc += pl[c] * mw[((size_t)(c + DD) * 3 + ky) * 3 + kx];
            }
        }
    }
    cert[idx] = 1.0f / (1.0f + expf(-acc));
}

// 8x8 block average of certainty (one block per batch, 64 threads)
__global__ void k_pool8(const float* __restrict__ cert, float* __restrict__ pooled) {
    int n = blockIdx.x;
    int t = threadIdx.x;
    int by = t >> 3, bx = t & 7;
    float s = 0.0f;
    for (int iy = 0; iy < 8; ++iy)
        for (int ix = 0; ix < 8; ++ix)
            s += cert[(size_t)n * HWH + (by * 8 + iy) * HH + bx * 8 + ix];
    pooled[n * 64 + t] = s * (1.0f / 64.0f);
}

// align_corners bilinear 8x8 -> 64x64
__global__ void k_avg_grid(const float* __restrict__ pooled, float* __restrict__ avg,
                           int total) {
    int idx = blockIdx.x * blockDim.x + threadIdx.x;
    if (idx >= total) return;
    int n = idx / HWH;
    int pix = idx % HWH;
    int oy = pix >> 6, ox = pix & 63;
    float fy = oy * (7.0f / 63.0f), fx = ox * (7.0f / 63.0f);
    int y0 = (int)floorf(fy), x0 = (int)floorf(fx);
    int y1 = min(y0 + 1, 7), x1 = min(x0 + 1, 7);
    float wy = fy - y0, wx = fx - x0;
    const float* pgrid = pooled + n * 64;
    avg[idx] = pgrid[y0 * 8 + x0] * (1 - wy) * (1 - wx) + pgrid[y0 * 8 + x1] * (1 - wy) * wx +
               pgrid[y1 * 8 + x0] * wy * (1 - wx)       + pgrid[y1 * 8 + x1] * wy * wx;
}

// x_high_edge = x_high * (1 - avg_grid) -> NHWC bf16
__global__ void k_edge_input(const float* __restrict__ xh, const float* __restrict__ avg,
                             __bf16* __restrict__ xedge, int total) {
    int idx = blockIdx.x * blockDim.x + threadIdx.x;
    if (idx >= total) return;
    int c   = idx & 255;
    int pix = (idx >> 8) & (HWH - 1);
    int n   = idx >> 20;
    float a = avg[n * HWH + pix];
    float v = xh[((size_t)(n * CP + c)) * HWH + pix];
    xedge[((size_t)(n * HWH + pix)) * CP + c] = (__bf16)(v * (1.0f - a));
}

// ef2: 3x3 conv 256 -> 1 over h1relu (NHWC bf16), output into d_out tail
__global__ void k_ef2(const __bf16* __restrict__ h1, const float* __restrict__ w2,
                      float* __restrict__ edge, int total) {
    int idx = blockIdx.x * blockDim.x + threadIdx.x;
    if (idx >= total) return;
    int n = idx / HWH;
    int pix = idx % HWH;
    int y = pix >> 6, x = pix & 63;
    float acc = 0.0f;
    for (int ky = 0; ky < 3; ++ky) {
        int yy = y + ky - 1;
        if (yy < 0 || yy >= HH) continue;
        for (int kx = 0; kx < 3; ++kx) {
            int xx = x + kx - 1;
            if (xx < 0 || xx >= HH) continue;
            const float* wk = w2 + (size_t)ky * 3 + kx; // stride 9 per channel
            const __bf16* ph = h1 + ((size_t)(n * HWH + yy * HH + xx)) * CP;
            for (int c = 0; c < CP; ++c)
                acc += (float)ph[c] * wk[(size_t)c * 9];
        }
    }
    edge[idx] = acc;
}

// per-batch top-P selection + coords + scatter indices (8 blocks x 256)
__global__ void k_topk(const float* __restrict__ edge, float* __restrict__ coords,
                       int* __restrict__ lowidx) {
    __shared__ float s[HWH];
    __shared__ float rmax[256];
    __shared__ int   ridx[256];
    int n = blockIdx.x, tid = threadIdx.x;
    for (int i = tid; i < HWH; i += 256) s[i] = edge[(size_t)n * HWH + i];
    __syncthreads();
    for (int it = 0; it < PP; ++it) {
        float m = -3.4e38f; int mi = 0;
        for (int i = tid; i < HWH; i += 256) {
            float v = s[i];
            if (v > m) { m = v; mi = i; }
        }
        rmax[tid] = m; ridx[tid] = mi;
        __syncthreads();
        for (int o = 128; o > 0; o >>= 1) {
            if (tid < o && rmax[tid + o] > rmax[tid]) {
                rmax[tid] = rmax[tid + o]; ridx[tid] = ridx[tid + o];
            }
            __syncthreads();
        }
        if (tid == 0) {
            int w = ridx[0];
            int px = w & 63, py = w >> 6;
            coords[((size_t)n * PP + it) * 2 + 0] = (px + 0.5f) / 64.0f;
            coords[((size_t)n * PP + it) * 2 + 1] = (py + 0.5f) / 64.0f;
            lowidx[n * PP + it] = px * 2 + py * 2 * HL;
            s[w] = -3.4e38f;
        }
        __syncthreads();
    }
}

// bilinear point gather of x_high and (PEG-convolved) x_low
__global__ void k_point_sample(const float* __restrict__ xh, const float* __restrict__ xl,
                               const float* __restrict__ pegw, const float* __restrict__ pegb,
                               const float* __restrict__ coords,
                               float* __restrict__ hf, float* __restrict__ lf) {
    int np = blockIdx.x;          // n*128 + p
    int n = np >> 7, p = np & 127;
    int c = threadIdx.x;
    float cx = coords[(size_t)np * 2 + 0];
    float cy = coords[(size_t)np * 2 + 1];
    // --- high-res gather (zeros padding, align_corners=False) ---
    {
        float x = cx * 64.0f - 0.5f, y = cy * 64.0f - 0.5f;
        float x0f = floorf(x), y0f = floorf(y);
        float wx = x - x0f, wy = y - y0f;
        int x0 = (int)x0f, y0 = (int)y0f;
        const float* img = xh + ((size_t)(n * CP + c)) * HWH;
        float acc = 0.0f;
#pragma unroll
        for (int t = 0; t < 4; ++t) {
            int yi = y0 + (t >> 1), xi = x0 + (t & 1);
            float wgt = ((t >> 1) ? wy : 1.0f - wy) * ((t & 1) ? wx : 1.0f - wx);
            float v = 0.0f;
            if (xi >= 0 && xi < HH && yi >= 0 && yi < HH) v = img[yi * HH + xi];
            acc += v * wgt;
        }
        hf[((size_t)(n * CP + c)) * PP + p] = acc;
    }
    // --- low-res gather with on-the-fly depthwise PEG conv + residual ---
    {
        float w9[9];
#pragma unroll
        for (int i = 0; i < 9; ++i) w9[i] = pegw[c * 9 + i];
        float bb = pegb[c];
        float x = cx * 128.0f - 0.5f, y = cy * 128.0f - 0.5f;
        float x0f = floorf(x), y0f = floorf(y);
        float wx = x - x0f, wy = y - y0f;
        int x0 = (int)x0f, y0 = (int)y0f;
        const float* img = xl + ((size_t)(n * CP + c)) * HWL;
        float acc = 0.0f;
#pragma unroll
        for (int t = 0; t < 4; ++t) {
            int yi = y0 + (t >> 1), xi = x0 + (t & 1);
            float wgt = ((t >> 1) ? wy : 1.0f - wy) * ((t & 1) ? wx : 1.0f - wx);
            float v = 0.0f;
            if (xi >= 0 && xi < HL && yi >= 0 && yi < HL) {
                float conv = 0.0f;
                for (int iy = 0; iy < 3; ++iy) {
                    int yy = yi + iy - 1;
                    if (yy < 0 || yy >= HL) continue;
                    for (int ix = 0; ix < 3; ++ix) {
                        int xx = xi + ix - 1;
                        if (xx < 0 || xx >= HL) continue;
                        conv += w9[iy * 3 + ix] * img[yy * HL + xx];
                    }
                }
                v = img[yi * HL + xi] + conv + bb;   // conv + bias + residual
            }
            acc += v * wgt;
        }
        lf[((size_t)(n * CP + c)) * PP + p] = acc;
    }
}

// logits[p][q] = sum_c low[c][p] * high[c][q]   (WMMA, K=256)
__global__ void k_aff_logits(const float* __restrict__ lfeat, const float* __restrict__ hfeat,
                             float* __restrict__ logits) {
    int gt = blockIdx.x * blockDim.x + threadIdx.x;
    int wid = gt >> 5, lane = gt & 31;
    int n = wid >> 6;
    int t = wid & 63;
    int p0 = (t >> 3) << 4, q0 = (t & 7) << 4;
    int lm = lane & 15, half = lane >> 4;
    const float* LF = lfeat + (size_t)n * CP * PP;
    const float* HF = hfeat + (size_t)n * CP * PP;
    v8f acc = {};
    for (int kb = 0; kb < 8; ++kb) {
        int c0 = kb * 32;
        v16bf a, b;
#pragma unroll
        for (int j = 0; j < 16; ++j) {
            int ka = half * 16 + j;
            a[j] = (__bf16)LF[(size_t)(c0 + ka) * PP + p0 + lm];
            int kq = half * 16 + j;
            b[j] = (__bf16)HF[(size_t)(c0 + kq) * PP + q0 + lm];
        }
        acc = __builtin_amdgcn_wmma_f32_16x16x32_bf16(false, a, false, b,
                                                      (short)0, acc, false, false);
    }
    float* L = logits + (size_t)n * PP * PP;
#pragma unroll
    for (int r = 0; r < 8; ++r) {
        int p = p0 + (half ? (r + 8) : r);
        L[(size_t)p * PP + q0 + lm] = acc[r];
    }
}

// row softmax over q (one block of 128 per row)
__global__ void k_softmax(float* __restrict__ logits) {
    __shared__ float red[PP];
    int row = blockIdx.x;
    int q = threadIdx.x;
    float* r = logits + (size_t)row * PP;
    float v = r[q];
    red[q] = v; __syncthreads();
    for (int o = 64; o > 0; o >>= 1) { if (q < o) red[q] = fmaxf(red[q], red[q + o]); __syncthreads(); }
    float mx = red[0]; __syncthreads();
    float e = expf(v - mx);
    red[q] = e; __syncthreads();
    for (int o = 64; o > 0; o >>= 1) { if (q < o) red[q] += red[q + o]; __syncthreads(); }
    r[q] = e / red[0];
}

// fusion[c][p] = sum_q aff[p][q]*high[c][q] + low[c][p]   (WMMA, K=128)
__global__ void k_aff_apply(const float* __restrict__ aff, const float* __restrict__ hfeat,
                            const float* __restrict__ lfeat, float* __restrict__ fusion) {
    int gt = blockIdx.x * blockDim.x + threadIdx.x;
    int wid = gt >> 5, lane = gt & 31;
    int n = wid >> 7;
    int t = wid & 127;
    int p0 = (t >> 4) << 4;
    int cb = (t & 15) << 4;
    int lm = lane & 15, half = lane >> 4;
    const float* AF = aff + (size_t)n * PP * PP;
    const float* HF = hfeat + (size_t)n * CP * PP;
    const float* LFn = lfeat + (size_t)n * CP * PP;
    v8f acc = {};
    for (int kb = 0; kb < 4; ++kb) {
        int q0 = kb * 32;
        v16bf a, b;
#pragma unroll
        for (int j = 0; j < 16; ++j) {
            int ka = half * 16 + j;
            a[j] = (__bf16)AF[(size_t)(p0 + lm) * PP + q0 + ka];
            int kq = half * 16 + j;
            b[j] = (__bf16)HF[(size_t)(cb + lm) * PP + q0 + kq];
        }
        acc = __builtin_amdgcn_wmma_f32_16x16x32_bf16(false, a, false, b,
                                                      (short)0, acc, false, false);
    }
    int c = cb + lm;
    float* F = fusion + (size_t)n * CP * PP;
#pragma unroll
    for (int r = 0; r < 8; ++r) {
        int p = p0 + (half ? (r + 8) : r);
        F[(size_t)c * PP + p] = acc[r] + LFn[(size_t)c * PP + p];
    }
}

// vectorized x_low -> out copy
__global__ void k_copy_f4(const float4* __restrict__ in, float4* __restrict__ out, int n4) {
    int i = blockIdx.x * blockDim.x + threadIdx.x;
    if (i < n4) out[i] = in[i];
}

// scatter fusion columns into the output grid
__global__ void k_scatter(const float* __restrict__ fusion, const int* __restrict__ lowidx,
                          float* __restrict__ out) {
    int np = blockIdx.x;          // n*128 + p
    int n = np >> 7, p = np & 127;
    int c = threadIdx.x;
    int li = lowidx[np];
    out[((size_t)(n * CP + c)) * HWL + li] = fusion[((size_t)(n * CP + c)) * PP + p];
}

// ------------------------------- launcher ----------------------------------
extern "C" void kernel_launch(void* const* d_in, const int* in_sizes, int n_in,
                              void* d_out, int out_size, void* d_ws, size_t ws_size,
                              hipStream_t stream) {
    (void)in_sizes; (void)n_in; (void)out_size; (void)ws_size;
    const float* x_high  = (const float*)d_in[0];
    const float* x_low   = (const float*)d_in[1];
    const float* down_h_w = (const float*)d_in[2];
    const float* down_h_b = (const float*)d_in[3];
    const float* down_l_w = (const float*)d_in[4];
    const float* down_l_b = (const float*)d_in[5];
    const float* match_w  = (const float*)d_in[6];
    const float* match_b  = (const float*)d_in[7];
    const float* ef1_w    = (const float*)d_in[8];
    const float* bn_g     = (const float*)d_in[9];
    const float* bn_b     = (const float*)d_in[10];
    const float* bn_rm    = (const float*)d_in[11];
    const float* bn_rv    = (const float*)d_in[12];
    const float* ef2_w    = (const float*)d_in[13];
    const float* peg_w    = (const float*)d_in[14];
    const float* peg_b    = (const float*)d_in[15];

    float* out      = (float*)d_out;
    float* edge_out = out + FINAL_ELEMS;   // (8,1,64,64) tail

    // workspace bump allocator (~175 MB total)
    char* base = (char*)d_ws;
    size_t off = 0;
    auto alloc = [&](size_t bytes) -> void* {
        off = (off + 255) & ~(size_t)255;
        void* p = base + off;
        off += bytes;
        return p;
    };
    __bf16* xh_nhwc  = (__bf16*)alloc((size_t)BATCH * HWH * CP * 2);
    __bf16* xl_nhwc  = (__bf16*)alloc((size_t)BATCH * HWL * CP * 2);
    __bf16* bp_dh    = (__bf16*)alloc((size_t)4 * 8 * 32 * 16 * 2);
    __bf16* bp_dl    = (__bf16*)alloc((size_t)4 * 8 * 32 * 16 * 2);
    __bf16* bp_ef1   = (__bf16*)alloc((size_t)9 * 16 * 8 * 32 * 16 * 2);
    float*  xh_emb   = (float*)alloc((size_t)BATCH * HWH * DD * 4);
    float*  xl_emb   = (float*)alloc((size_t)BATCH * HWL * DD * 4);
    float*  xl_up    = (float*)alloc((size_t)BATCH * HWH * DD * 4);
    float*  cert     = (float*)alloc((size_t)BATCH * HWH * 4);
    float*  pooled   = (float*)alloc((size_t)BATCH * 64 * 4);
    float*  avg      = (float*)alloc((size_t)BATCH * HWH * 4);
    __bf16* xedge    = (__bf16*)alloc((size_t)BATCH * HWH * CP * 2);
    __bf16* h1relu   = (__bf16*)alloc((size_t)BATCH * HWH * CP * 2);
    float*  coords   = (float*)alloc((size_t)BATCH * PP * 2 * 4);
    int*    lowidx   = (int*)alloc((size_t)BATCH * PP * 4);
    float*  hfeat    = (float*)alloc((size_t)BATCH * CP * PP * 4);
    float*  lfeat    = (float*)alloc((size_t)BATCH * CP * PP * 4);
    float*  logits   = (float*)alloc((size_t)BATCH * PP * PP * 4);
    float*  fusion   = (float*)alloc((size_t)BATCH * CP * PP * 4);

    const int T = 256;
    // 1. NHWC bf16 activations
    {
        int tot = BATCH * CP * HWH;
        k_to_nhwc_bf16<<<(tot + T - 1) / T, T, 0, stream>>>(x_high, xh_nhwc, CP, HWH, tot);
    }
    {
        int tot = BATCH * CP * HWL;
        k_to_nhwc_bf16<<<(tot + T - 1) / T, T, 0, stream>>>(x_low, xl_nhwc, CP, HWL, tot);
    }
    // 2. packed weights
    k_pack_w<<<(16384 + T - 1) / T, T, 0, stream>>>(down_h_w, bp_dh, 64, 256, 1, 4, 8, 16384);
    k_pack_w<<<(16384 + T - 1) / T, T, 0, stream>>>(down_l_w, bp_dl, 64, 256, 1, 4, 8, 16384);
    k_pack_w<<<(589824 + T - 1) / T, T, 0, stream>>>(ef1_w, bp_ef1, 256, 256, 9, 16, 8, 589824);
    // 3-4. embedding 1x1 convs (WMMA GEMM, 16x64 tile per wave)
    k_gemm_1x1<<<(2048 * 32) / T, T, 0, stream>>>(xh_nhwc, bp_dh, down_h_b, xh_emb, 2048);
    k_gemm_1x1<<<(8192 * 32) / T, T, 0, stream>>>(xl_nhwc, bp_dl, down_l_b, xl_emb, 8192);
    // 5. upsample low embedding 128->64 (align_corners)
    {
        int tot = BATCH * HWH * DD;
        k_upsample_ac<<<(tot + T - 1) / T, T, 0, stream>>>(xl_emb, xl_up, DD, HL, HL, HH, HH, tot);
    }
    // 6. point matcher conv + sigmoid
    k_match<<<(BATCH * HWH + T - 1) / T, T, 0, stream>>>(xh_emb, xl_up, match_w, match_b,
                                                        cert, BATCH * HWH);
    // 7-8. 8x8 avg pool, bilinear back up, form x_high_edge (NHWC bf16)
    k_pool8<<<BATCH, 64, 0, stream>>>(cert, pooled);
    k_avg_grid<<<(BATCH * HWH + T - 1) / T, T, 0, stream>>>(pooled, avg, BATCH * HWH);
    {
        int tot = BATCH * CP * HWH;
        k_edge_input<<<(tot + T - 1) / T, T, 0, stream>>>(x_high, avg, xedge, tot);
    }
    // 9. ef1 3x3 conv + BN + ReLU (WMMA GEMM, async-LDS staged weights)
    k_gemm_ef1<<<256 * 16, T, 0, stream>>>(xedge, bp_ef1, bn_g, bn_b, bn_rm, bn_rv, h1relu);
    // 10. ef2 -> edge_pred (written straight into d_out tail)
    k_ef2<<<(BATCH * HWH + T - 1) / T, T, 0, stream>>>(h1relu, ef2_w, edge_out, BATCH * HWH);
    // 11. top-128 points per batch
    k_topk<<<BATCH, 256, 0, stream>>>(edge_out, coords, lowidx);
    // 12. point sampling (x_high + PEG'd x_low)
    k_point_sample<<<BATCH * PP, CP, 0, stream>>>(x_high, x_low, peg_w, peg_b, coords,
                                                  hfeat, lfeat);
    // 13-15. attention: logits (WMMA) -> softmax -> apply (WMMA) + residual
    k_aff_logits<<<(512 * 32) / T, T, 0, stream>>>(lfeat, hfeat, logits);
    k_softmax<<<BATCH * PP, PP, 0, stream>>>(logits);
    k_aff_apply<<<(1024 * 32) / T, T, 0, stream>>>(logits, hfeat, lfeat, fusion);
    // 16-17. assemble final features: copy x_low then scatter fused columns
    {
        int n4 = (int)(FINAL_ELEMS / 4);
        k_copy_f4<<<(n4 + T - 1) / T, T, 0, stream>>>((const float4*)x_low, (float4*)out, n4);
    }
    k_scatter<<<BATCH * PP, CP, 0, stream>>>(fusion, lowidx, out);
}
